// CrossAttention_2808908612006
// MI455X (gfx1250) — compile-verified, matrix operand order checked by reading, not run on hardware
//
#include <hip/hip_runtime.h>
#include <math.h>

// ---------------------------------------------------------------------------
// CrossAttention for MI455X (gfx1250, wave32, WMMA 16x16x32 f16)
// Q = tokens @ Wq ; K = context @ Wk ; V = context @ Wv (stored transposed)
// O = softmax(Q K^T / sqrt(EMB)) V
// ---------------------------------------------------------------------------

typedef __attribute__((ext_vector_type(16))) _Float16 v16h;
typedef __attribute__((ext_vector_type(8)))  _Float16 v8h;
typedef __attribute__((ext_vector_type(4)))  _Float16 v4h;
typedef __attribute__((ext_vector_type(8)))  float    v8f;

#define B_    8
#define T_    4096
#define S_    1024
#define HID_  512
#define EMB_  512
#define CTX_  768

#define PSTR  1032   // f16 score-row stride in LDS: (1032*2)B = 129*16B -> 4-bank stagger
#define VSTR  40     // f16 V-slice row stride in LDS (conflict-free b128 fragments)

// Load one 16x32 f16 A/B fragment row for this lane.
// Layout (CDNA5 ISA 7.12.2): lanes 0-15 -> k in [0..7]∪[16..23], lanes 16-31 -> +8.
// Caller passes kb = kstep + ((lane>=16)?8:0); rowPtr = start of this lane's row.
__device__ __forceinline__ v16h ldfrag16(const _Float16* rowPtr, int kb) {
  v8h lo = *(const v8h*)(rowPtr + kb);
  v8h hi = *(const v8h*)(rowPtr + kb + 16);
  v16h r;
#pragma unroll
  for (int i = 0; i < 8; ++i) { r[i] = lo[i]; r[i + 8] = hi[i]; }
  return r;
}

// Async global->LDS 16-byte copy (CDNA5 GLOBAL_LOAD_ASYNC_TO_LDS_B128, ASYNCcnt).
// lds_off = wave-relative LDS byte address (low 32 bits of generic shared ptr).
__device__ __forceinline__ void async_ld_b128(unsigned lds_off, const void* gaddr) {
  unsigned long long ga = (unsigned long long)(uintptr_t)gaddr;
  asm volatile("global_load_async_to_lds_b128 %0, %1, off"
               :: "v"(lds_off), "v"(ga) : "memory");
}

// ---------------------------------------------------------------------------
// Projection GEMM: Out_f16[M][N] = A_f32[M][K] * W_f32[K][N]
// TS == 0 : row-major f16 output
// TS  > 0 : per-batch transposed output Vt[b][n][s], s = m % TS (TS = power of 2)
// Block: 256 threads (8 waves as 4Mx2N), tile 128(M) x 128(N), k-step 32,
// register double-buffered staging (global loads overlap WMMA).
// ---------------------------------------------------------------------------
template <int TS>
__global__ __launch_bounds__(256) void proj_gemm(
    const float* __restrict__ A, const float* __restrict__ W,
    _Float16* __restrict__ Out, int M, int K, int N) {

  __shared__ _Float16 As[128 * 40];  // 128 m-rows x 32 k, stride 40 (conflict-free b128)
  __shared__ _Float16 Bs[128 * 40];  // 128 n-rows (transposed W) x 32 k, stride 40

  const int tid  = threadIdx.x;
  const int lane = tid & 31;
  const int wave = tid >> 5;
  const int wm   = wave & 3;      // 4 waves along M (32 rows each)
  const int wn   = wave >> 2;     // 2 waves along N (64 cols each)
  const int m0   = blockIdx.y * 128;
  const int n0   = blockIdx.x * 128;

  const int l15  = lane & 15;
  const int koff = (lane >= 16) ? 8 : 0;

  auto loadA = [&](int k0, float4* pa) {
#pragma unroll
    for (int it = 0; it < 4; ++it) {
      int idx = tid + it * 256;            // 0..1023
      int r   = idx >> 3;                  // m-row 0..127
      int c4  = idx & 7;                   // float4 group 0..7
      pa[it] = *(const float4*)(A + (size_t)(m0 + r) * K + k0 + c4 * 4);
    }
  };
  auto loadW = [&](int k0, float4* pw) {
#pragma unroll
    for (int it = 0; it < 4; ++it) {
      int idx = tid + it * 256;            // 0..1023
      int kr  = idx >> 5;                  // k-row 0..31
      int c4  = idx & 31;                  // n group 0..31
      pw[it] = *(const float4*)(W + (size_t)(k0 + kr) * N + n0 + c4 * 4);
    }
  };
  auto storeTiles = [&](const float4* pa, const float4* pw) {
#pragma unroll
    for (int it = 0; it < 4; ++it) {
      int idx = tid + it * 256;
      int r   = idx >> 3;
      int c4  = idx & 7;
      v4h h = { (_Float16)pa[it].x, (_Float16)pa[it].y,
                (_Float16)pa[it].z, (_Float16)pa[it].w };
      *(v4h*)(As + r * 40 + c4 * 4) = h;
    }
#pragma unroll
    for (int it = 0; it < 4; ++it) {
      int idx = tid + it * 256;
      int kr  = idx >> 5;
      int c4  = idx & 31;
      Bs[(c4 * 4 + 0) * 40 + kr] = (_Float16)pw[it].x;
      Bs[(c4 * 4 + 1) * 40 + kr] = (_Float16)pw[it].y;
      Bs[(c4 * 4 + 2) * 40 + kr] = (_Float16)pw[it].z;
      Bs[(c4 * 4 + 3) * 40 + kr] = (_Float16)pw[it].w;
    }
  };

  float4 pa[4], pw[4];
  loadA(0, pa);
  loadW(0, pw);

  v8f acc[2][4] = {};

  for (int k0 = 0; k0 < K; k0 += 32) {
    storeTiles(pa, pw);
    __syncthreads();
    // Prefetch next k-slice while this slice's WMMAs run.
    if (k0 + 32 < K) { loadA(k0 + 32, pa); loadW(k0 + 32, pw); }

    v16h af[2], bf[4];
#pragma unroll
    for (int i = 0; i < 2; ++i)
      af[i] = ldfrag16(As + (wm * 32 + i * 16 + l15) * 40, koff);
#pragma unroll
    for (int j = 0; j < 4; ++j)
      bf[j] = ldfrag16(Bs + (wn * 64 + j * 16 + l15) * 40, koff);
#pragma unroll
    for (int i = 0; i < 2; ++i)
#pragma unroll
      for (int j = 0; j < 4; ++j)
        acc[i][j] = __builtin_amdgcn_wmma_f32_16x16x32_f16(
            false, af[i], false, bf[j], (short)0, acc[i][j], false, false);
    __syncthreads();
  }

  // Epilogue. C/D layout: lane holds column n = l15 of its 16x16 tile,
  // rows (koff + 0..7) in acc[0..7].
  constexpr int TSS = (TS == 0) ? 1 : TS;   // avoid div-by-zero in dead branch
#pragma unroll
  for (int i = 0; i < 2; ++i) {
#pragma unroll
    for (int j = 0; j < 4; ++j) {
      int gm = m0 + wm * 32 + i * 16 + koff;       // first of 8 rows
      int gn = n0 + wn * 64 + j * 16 + l15;
      if (TS == 0) {
#pragma unroll
        for (int e = 0; e < 8; ++e)
          Out[(size_t)(gm + e) * N + gn] = (_Float16)acc[i][j][e];
      } else {
        int bidx = gm / TSS;                       // compile-time shift
        int s    = gm - bidx * TSS;                // 8 consecutive s, same batch
        v8h h;
#pragma unroll
        for (int e = 0; e < 8; ++e) h[e] = (_Float16)acc[i][j][e];
        *(v8h*)(Out + ((size_t)bidx * N + gn) * TSS + s) = h;
      }
    }
  }
}

// ---------------------------------------------------------------------------
// Fused attention: block = (batch b, 64-row T tile), 256 threads = 8 waves.
// Pass1: scores (WMMA, K direct from global/L2) -> f16 LDS, row max in regs.
// Softmax: exp in place (deferred normalization).
// Pass2: O = P*V with Vt k-slices async-staged to LDS (double-buffered,
//        GLOBAL_LOAD_ASYNC_TO_LDS_B128 + s_wait_asynccnt).
// ---------------------------------------------------------------------------
__global__ __launch_bounds__(256) void attn_kernel(
    const _Float16* __restrict__ Q,   // [B][T][EMB] row-major
    const _Float16* __restrict__ Kt,  // [B][S][EMB] row-major (rows = B-fragments of K^T)
    const _Float16* __restrict__ Vt,  // [B][HID][S] (rows = B-fragments of V)
    float* __restrict__ Out) {        // [B][T][HID]

  extern __shared__ char smem_raw[];
  _Float16* P    = (_Float16*)smem_raw;        // [64][PSTR] f16 scores/probs
  _Float16* VtL0 = P + 64 * PSTR;              // [512][VSTR] V slice buf 0
  _Float16* VtL1 = VtL0 + 512 * VSTR;          // [512][VSTR] V slice buf 1
  float* maxwave = (float*)(VtL1 + 512 * VSTR);// [8][16]
  float* rowmax  = maxwave + 128;              // [64]
  float* partial = rowmax + 64;                // [64][4]
  float* rinv    = partial + 256;              // [64]

  const int tid  = threadIdx.x;
  const int lane = tid & 31;
  const int wave = tid >> 5;
  const int l15  = lane & 15;
  const int koff = (lane >= 16) ? 8 : 0;

  const int t0 = blockIdx.x * 64;
  const int b  = blockIdx.y;

  const _Float16* Qb = Q  + (size_t)b * T_   * EMB_;
  const _Float16* Kb = Kt + (size_t)b * S_   * EMB_;
  const _Float16* Vb = Vt + (size_t)b * HID_ * S_;

  const float alpha = 0.044194173824159216f;  // 1/sqrt(512)

  // ---- Pass 1: scores ----
  const int rg = wave >> 1;                  // row-group 0..3 (16 T-rows each)
  const int cg = wave & 1;                   // S half 0..1 (512 cols each)
  const _Float16* qrow = Qb + (size_t)(t0 + rg * 16 + l15) * EMB_;

  float rmax[8];
#pragma unroll
  for (int e = 0; e < 8; ++e) rmax[e] = -3.0e38f;

  for (int chunk = 0; chunk < 4; ++chunk) {
    const int cBase = cg * 512 + chunk * 128;
    v8f sacc[8] = {};
    for (int ks = 0; ks < EMB_; ks += 32) {
      const int kb = ks + koff;
      v16h aq = ldfrag16(qrow, kb);
#pragma unroll
      for (int j = 0; j < 8; ++j) {
        v16h bk = ldfrag16(Kb + (size_t)(cBase + j * 16 + l15) * EMB_, kb);
        sacc[j] = __builtin_amdgcn_wmma_f32_16x16x32_f16(
            false, aq, false, bk, (short)0, sacc[j], false, false);
      }
    }
    // scale, track row max, spill f16 scores to LDS
#pragma unroll
    for (int j = 0; j < 8; ++j) {
      const int n = cBase + j * 16 + l15;
#pragma unroll
      for (int e = 0; e < 8; ++e) {
        float v = sacc[j][e] * alpha;
        rmax[e] = fmaxf(rmax[e], v);
        P[(rg * 16 + koff + e) * PSTR + n] = (_Float16)v;
      }
    }
  }
  // reduce rmax within each 16-lane group (columns of the wave's tiles)
#pragma unroll
  for (int m = 1; m <= 8; m <<= 1)
#pragma unroll
    for (int e = 0; e < 8; ++e)
      rmax[e] = fmaxf(rmax[e], __shfl_xor(rmax[e], m, 32));
  if (l15 == 0) {
#pragma unroll
    for (int e = 0; e < 8; ++e)
      maxwave[wave * 16 + koff + e] = rmax[e];
  }
  __syncthreads();

  if (tid < 64) {
    int rrg = tid >> 4, li = tid & 15;
    rowmax[tid] = fmaxf(maxwave[(2 * rrg) * 16 + li],
                        maxwave[(2 * rrg + 1) * 16 + li]);
  }
  __syncthreads();

  // ---- softmax numerators in place (normalization deferred) ----
  {
    int r = tid >> 2;                 // row 0..63
    int q = tid & 3;                  // quarter of S
    float mx = rowmax[r];
    float s = 0.f;
    _Float16* prow = P + r * PSTR + q * 256;
#pragma unroll 4
    for (int c = 0; c < 256; ++c) {
      float e = __expf((float)prow[c] - mx);
      s += e;
      prow[c] = (_Float16)e;
    }
    partial[r * 4 + q] = s;
  }
  __syncthreads();
  if (tid < 64) {
    float s = partial[tid * 4 + 0] + partial[tid * 4 + 1] +
              partial[tid * 4 + 2] + partial[tid * 4 + 3];
    rinv[tid] = 1.0f / s;
  }
  __syncthreads();

  // ---- Pass 2: O = P * V, V slices async-staged to LDS ----
  const int org = wave >> 1;                // row-group 0..3
  const int hg  = wave & 1;                 // HID half 0..1 (256 cols)
  const int m0  = org * 16;
  const int h0  = hg * 256;

  _Float16* vbuf[2] = { VtL0, VtL1 };

  // Issue one 512x32 V k-slice: 2048 b128 copies, 8 per thread (+8 ASYNCcnt/wave).
  auto issue_slice = [&](int buf, int ks) {
    const _Float16* src = Vb + ks;
#pragma unroll
    for (int it = 0; it < 8; ++it) {
      int idx  = tid + it * 256;            // 0..2047
      int row  = idx >> 2;                  // h-row 0..511
      int part = idx & 3;                   // 16B chunk within 64B row segment
      unsigned loff = (unsigned)(uintptr_t)(vbuf[buf] + row * VSTR + part * 8);
      async_ld_b128(loff, src + (size_t)row * S_ + part * 8);
    }
  };

  issue_slice(0, 0);
  int cur = 0;

  v8f oacc[16] = {};
  const _Float16* prow = P + (size_t)(m0 + l15) * PSTR;
  for (int ks = 0; ks < S_; ks += 32) {
    if (ks + 32 < S_) {
      issue_slice(cur ^ 1, ks + 32);
      asm volatile("s_wait_asynccnt 0x8" ::: "memory");   // slice `cur` landed
    } else {
      asm volatile("s_wait_asynccnt 0x0" ::: "memory");
    }
    __syncthreads();   // make other waves' async-loaded data visible

    v16h ap = ldfrag16(prow, ks + koff);
    const _Float16* vb = vbuf[cur];
#pragma unroll
    for (int nt = 0; nt < 16; ++nt) {
      v16h bv = ldfrag16(vb + (h0 + nt * 16 + l15) * VSTR, koff);
      oacc[nt] = __builtin_amdgcn_wmma_f32_16x16x32_f16(
          false, ap, false, bv, (short)0, oacc[nt], false, false);
    }
    __syncthreads();   // protect buffer `cur` before it is overwritten
    cur ^= 1;
  }

  float sc[8];
#pragma unroll
  for (int e = 0; e < 8; ++e) sc[e] = rinv[m0 + koff + e];

  float* outb = Out + ((size_t)b * T_ + t0 + m0 + koff) * HID_;
#pragma unroll
  for (int nt = 0; nt < 16; ++nt) {
    int n = h0 + nt * 16 + l15;
#pragma unroll
    for (int e = 0; e < 8; ++e)
      outb[(size_t)e * HID_ + n] = oacc[nt][e] * sc[e];
  }
}

// ---------------------------------------------------------------------------
extern "C" void kernel_launch(void* const* d_in, const int* in_sizes, int n_in,
                              void* d_out, int out_size, void* d_ws, size_t ws_size,
                              hipStream_t stream) {
  const float* tokens  = (const float*)d_in[0];  // [B][T][HID]
  const float* context = (const float*)d_in[1];  // [B][S][CTX]
  const float* Wq      = (const float*)d_in[2];  // [HID][EMB]
  const float* Wk      = (const float*)d_in[3];  // [CTX][EMB]
  const float* Wv      = (const float*)d_in[4];  // [CTX][HID]
  float*       out     = (float*)d_out;          // [B][T][HID]

  // Workspace: Q f16 (32 MB) | K f16 (8 MB) | Vt f16 (8 MB)
  _Float16* Qh = (_Float16*)d_ws;
  _Float16* Kh = Qh + (size_t)B_ * T_ * EMB_;
  _Float16* Vt = Kh + (size_t)B_ * S_ * EMB_;

  dim3 blk(256);
  // Q = tokens @ Wq        : M=B*T=32768, K=512, N=512
  proj_gemm<0><<<dim3(EMB_ / 128, (B_ * T_) / 128), blk, 0, stream>>>(
      tokens, Wq, Qh, B_ * T_, HID_, EMB_);
  // K = context @ Wk       : M=B*S=8192, K=768, N=512
  proj_gemm<0><<<dim3(EMB_ / 128, (B_ * S_) / 128), blk, 0, stream>>>(
      context, Wk, Kh, B_ * S_, CTX_, EMB_);
  // V = context @ Wv, stored transposed per batch: Vt[b][h][s]
  proj_gemm<S_><<<dim3(HID_ / 128, (B_ * S_) / 128), blk, 0, stream>>>(
      context, Wv, Vt, B_ * S_, CTX_, HID_);

  size_t smem = (size_t)64 * PSTR * 2            // P
              + (size_t)2 * 512 * VSTR * 2       // V slice double buffer
              + (size_t)(128 + 64 + 256 + 64) * sizeof(float);
  (void)hipFuncSetAttribute((const void*)attn_kernel,
                            hipFuncAttributeMaxDynamicSharedMemorySize, (int)smem);
  attn_kernel<<<dim3(T_ / 64, B_), blk, smem, stream>>>(Qh, Kh, Vt, out);
}
